// PcConv_28767690948914
// MI455X (gfx1250) — matrix operand order repaired and checked by reading, not compiled, not run on hardware
//
#include <hip/hip_runtime.h>

typedef __attribute__((ext_vector_type(16))) _Float16 v16h;
typedef __attribute__((ext_vector_type(8)))  _Float16 v8h;
typedef __attribute__((ext_vector_type(8)))  float    v8f;

#define B_    8
#define CIN   3
#define NPTS  4096
#define KNB   32
#define CF    64
#define C1_   64
#define C2_   128
#define TN    64
#define EPS_  1e-3f

// colsum[b*N+n] = (1/K) * sum_c feat[b,c,n]
__global__ __launch_bounds__(256) void colsum_kernel(const float* __restrict__ feat,
                                                     float* __restrict__ colsum) {
  int j = blockIdx.x * 256 + threadIdx.x;            // [0, B*N)
  int b = j >> 12, n = j & (NPTS - 1);
  const float* fp = feat + ((size_t)b * CF) * NPTS + n;
  float s = 0.f;
#pragma unroll
  for (int c = 0; c < CF; ++c) s += fp[(size_t)c * NPTS];
  colsum[j] = s * (1.0f / KNB);
}

__global__ __launch_bounds__(256) void pcconv_kernel(
    const float* __restrict__ xyz, const int* __restrict__ ind,
    const float* __restrict__ W1, const float* __restrict__ b1,
    const float* __restrict__ g1, const float* __restrict__ be1,
    const float* __restrict__ m1, const float* __restrict__ v1,
    const float* __restrict__ W2, const float* __restrict__ b2,
    const float* __restrict__ g2, const float* __restrict__ be2,
    const float* __restrict__ m2, const float* __restrict__ v2,
    const float* __restrict__ colsum, float* __restrict__ out) {
  __shared__ __align__(16) _Float16 h1T[KNB][72];    // [neighbor k][c1], row padded to 144B
  __shared__ float sv[KNB];
  __shared__ float stage[C2_ * TN];

  const int tid = threadIdx.x;
  const int w   = tid >> 5;          // wave id 0..7 (wave32)
  const int l   = tid & 31;
  const int lh  = l >> 4;            // half-wave select
  const int lm  = l & 15;

  const int b  = blockIdx.x / (NPTS / TN);
  const int n0 = (blockIdx.x % (NPTS / TN)) * TN;

  // ---- layer-1 A fragment: BN folded into W1, bias as K=3 column, zero-padded K ----
  const int c1i = 16 * (w >> 1) + lm;
  float s1  = g1[c1i] * rsqrtf(v1[c1i] + EPS_);
  float bf1 = (b1[c1i] - m1[c1i]) * s1 + be1[c1i];
  v16h a1f;
#pragma unroll
  for (int j = 0; j < 16; ++j) {
    int Kj = (j < 8) ? (8 * lh + j) : (16 + 8 * lh + (j - 8));  // ISA 16-bit A layout
    float val = 0.f;
    if (Kj < CIN)       val = W1[c1i * CIN + Kj] * s1;
    else if (Kj == CIN) val = bf1;
    a1f[j] = (_Float16)val;
  }

  // ---- layer-2 A fragments: W2 rows [16w,16w+16), K halves 0..31 / 32..63 ----
  const int o2 = 16 * w + lm;
  v16h a2_0, a2_1;
#pragma unroll
  for (int j = 0; j < 16; ++j) {
    int Kj = 16 * lh + j;
    a2_0[j] = (_Float16)W2[o2 * C1_ + Kj];
    a2_1[j] = (_Float16)W2[o2 * C1_ + 32 + Kj];
  }

  // ---- layer-2 BN epilogue constants for this lane's 8 output rows ----
  float s2r[8], b2r[8];
#pragma unroll
  for (int r = 0; r < 8; ++r) {
    int o = 16 * w + r + 8 * lh;
    float s2 = g2[o] * rsqrtf(v2[o] + EPS_);
    s2r[r] = s2;
    b2r[r] = (b2[o] - m2[o]) * s2 + be2[o];
  }

  const int tk   = w & 1;            // layer-1 neighbor-tile for this wave
  const int kcol = 16 * tk + lm;

  for (int jp = 0; jp < TN; ++jp) {
    const int n = n0 + jp;

    if (w == 0) {                    // gathered channel-sums (1/K already folded in)
      int idx = ind[((size_t)b * NPTS + n) * KNB + l];
      sv[l] = colsum[b * NPTS + idx];
    }

    // ---- layer 1: H1 = relu([W1f|b1f] @ [x;1]) -> one WMMA per wave ----
    v16h bx;
#pragma unroll
    for (int j = 0; j < 16; ++j) bx[j] = (_Float16)0.f;
    if (lh == 0) {                   // B rows K=0..3 live in lanes 0-15 (K=j)
      const float* xp = xyz + (((size_t)b * CIN) * NPTS + n) * KNB + kcol;
      bx[0] = (_Float16)xp[0];
      bx[1] = (_Float16)xp[(size_t)NPTS * KNB];
      bx[2] = (_Float16)xp[2 * (size_t)NPTS * KNB];
      bx[3] = (_Float16)1.0f;
    }
    v8f c1acc = {};
    c1acc = __builtin_amdgcn_wmma_f32_16x16x32_f16(false, a1f, false, bx,
                                                   (short)0, c1acc, false, false);
    v8h hh;
#pragma unroll
    for (int r = 0; r < 8; ++r) hh[r] = (_Float16)fmaxf(c1acc[r], 0.f);
    *(v8h*)&h1T[kcol][16 * (w >> 1) + 8 * lh] = hh;   // h1T[k][c1], 16B store

    __syncthreads();

    // ---- layer 2: [16x64] @ [64x32] per wave = 4 WMMAs ----
    v16h bfr[2][2];
#pragma unroll
    for (int t = 0; t < 2; ++t)
#pragma unroll
      for (int s = 0; s < 2; ++s) {
        const v8h* bp = (const v8h*)&h1T[16 * t + lm][32 * s + 16 * lh];
        v8h lo = bp[0], hi = bp[1];  // two ds_load_b128 per fragment
        bfr[t][s] = __builtin_shufflevector(lo, hi,
            0,1,2,3,4,5,6,7,8,9,10,11,12,13,14,15);
      }
    v8f acc0 = {}, acc1 = {};
    acc0 = __builtin_amdgcn_wmma_f32_16x16x32_f16(false, a2_0, false, bfr[0][0], (short)0, acc0, false, false);
    acc0 = __builtin_amdgcn_wmma_f32_16x16x32_f16(false, a2_1, false, bfr[0][1], (short)0, acc0, false, false);
    acc1 = __builtin_amdgcn_wmma_f32_16x16x32_f16(false, a2_0, false, bfr[1][0], (short)0, acc1, false, false);
    acc1 = __builtin_amdgcn_wmma_f32_16x16x32_f16(false, a2_1, false, bfr[1][1], (short)0, acc1, false, false);

    // ---- epilogue: relu(BN) * sv, reduce over 32 neighbors ----
    float svA = sv[lm], svB = sv[16 + lm];
    float part[8];
#pragma unroll
    for (int r = 0; r < 8; ++r) {
      float w0 = fmaxf(acc0[r] * s2r[r] + b2r[r], 0.f);
      float w1 = fmaxf(acc1[r] * s2r[r] + b2r[r], 0.f);
      part[r] = w0 * svA + w1 * svB;
    }
#pragma unroll
    for (int m = 1; m < 16; m <<= 1) {
#pragma unroll
      for (int r = 0; r < 8; ++r) part[r] += __shfl_xor(part[r], m, 32);
    }
    if (lm == 0) {
#pragma unroll
      for (int r = 0; r < 8; ++r)
        stage[(16 * w + r + 8 * lh) * TN + jp] = part[r];
    }
    __syncthreads();                 // h1T/sv reuse + stage visibility
  }

  // ---- coalesced writeout of the [128 x TN] output tile ----
#pragma unroll 4
  for (int i = tid; i < C2_ * TN; i += 256) {
    int o = i / TN, j = i % TN;
    out[((size_t)b * C2_ + o) * NPTS + n0 + j] = stage[i];
  }
}

extern "C" void kernel_launch(void* const* d_in, const int* in_sizes, int n_in,
                              void* d_out, int out_size, void* d_ws, size_t ws_size,
                              hipStream_t stream) {
  const float* xyz  = (const float*)d_in[0];
  const float* feat = (const float*)d_in[1];
  const int*   ind  = (const int*)d_in[2];
  const float* W1   = (const float*)d_in[3];
  const float* b1   = (const float*)d_in[4];
  const float* g1   = (const float*)d_in[5];
  const float* be1  = (const float*)d_in[6];
  const float* m1   = (const float*)d_in[7];
  const float* v1   = (const float*)d_in[8];
  const float* W2   = (const float*)d_in[9];
  const float* b2   = (const float*)d_in[10];
  const float* g2   = (const float*)d_in[11];
  const float* be2  = (const float*)d_in[12];
  const float* m2   = (const float*)d_in[13];
  const float* v2   = (const float*)d_in[14];
  float* colsum = (float*)d_ws;                       // 128 KB scratch
  float* out    = (float*)d_out;

  colsum_kernel<<<(B_ * NPTS) / 256, 256, 0, stream>>>(feat, colsum);
  pcconv_kernel<<<B_ * (NPTS / TN), 256, 0, stream>>>(
      xyz, ind, W1, b1, g1, be1, m1, v1, W2, b2, g2, be2, m2, v2, colsum, out);
}
